// MinGRU_2834678415394
// MI455X (gfx1250) — matrix operand order repaired and checked by reading, not compiled, not run on hardware
//
#include <hip/hip_runtime.h>
#include <hip/hip_bf16.h>
#include <math.h>

// ---- problem constants (from setup_inputs) ----
#define B_SZ      4
#define T_SEQ     4096
#define D_DIM     1024
#define M_TOT     (B_SZ * T_SEQ)     // 16384 rows of the fused GEMM
#define CHUNKS    64
#define CHUNK_LEN (T_SEQ / CHUNKS)   // 64

typedef __attribute__((ext_vector_type(16))) _Float16 v16h;
typedef __attribute__((ext_vector_type(8)))  _Float16 v8h;
typedef __attribute__((ext_vector_type(4)))  _Float16 h4;
typedef __attribute__((ext_vector_type(8)))  float    v8f;

// ---------- math helpers ----------
__device__ __forceinline__ float softplus_f(float x) {
    // stable softplus: max(x,0) + log1p(exp(-|x|))
    return fmaxf(x, 0.f) + log1pf(__expf(-fabsf(x)));
}
__device__ __forceinline__ float logaddexp_f(float x, float y) {
    float m = fmaxf(x, y);
    return m + log1pf(__expf(-fabsf(x - y)));
}
__device__ __forceinline__ float log_g_f(float x) {
    return (x >= 0.f) ? __logf(x + 0.5f) : -softplus_f(-x);
}

// =====================================================================
// Kernel 0: fp32 -> f16 conversion pass (one shot; removes all in-loop
// cvt work and halves GEMM load bytes). Vectorized float4 -> half4.
// =====================================================================
__global__ __launch_bounds__(256)
void cvt_f32_to_f16(const float* __restrict__ in, _Float16* __restrict__ out)
{
    size_t i = (size_t)blockIdx.x * blockDim.x + threadIdx.x;  // one float4 per thread
    float4 f = ((const float4*)in)[i];
    h4 h;
    h[0] = (_Float16)f.x; h[1] = (_Float16)f.y;
    h[2] = (_Float16)f.z; h[3] = (_Float16)f.w;
    ((h4*)out)[i] = h;
}

// ---------- f16 fragment loaders (contiguous k = 16*half + s order, same for A and B) ----------
// A-matrix 16x32 f16: lane<16 holds K = 0..15; lane>=16 holds K = 16..31.
// Caller passes p = row_ptr + k0 + half*16; elements 0..15 <- p[0..15].
__device__ __forceinline__ v16h load_a16(const _Float16* __restrict__ p) {
    v8h lo = *(const v8h*)(p);        // 16B
    v8h hi = *(const v8h*)(p + 8);    // 16B
    return __builtin_shufflevector(lo, hi, 0,1,2,3,4,5,6,7,8,9,10,11,12,13,14,15);
}
// B-matrix 32x16 f16: lane<16 column n, K=0..15; lane>=16 column n, K=16..31.
// B[k][n] = W[n][k] -> 16 consecutive halfs of W row n at k0 + half*16 (32B aligned).
__device__ __forceinline__ v16h load_b16(const _Float16* __restrict__ p) {
    return *(const v16h*)(p);
}

// ---------- epilogue: gate math + store, fully unrolled, no arrays ----------
__device__ __forceinline__ void store_gate_tile(v8f accK, v8f accP,
                                                int m0, int n0, int half, int l16,
                                                float* __restrict__ lc,
                                                float* __restrict__ lv)
{
    const int col = n0 + l16;
    #pragma unroll
    for (int j = 0; j < 8; ++j) {
        int row = m0 + half * 8 + j;          // C/D layout: VGPR j -> M=j (+8 for hi half)
        float kv = accK[j];
        float pv = accP[j];
        float log_z = -softplus_f(-kv);       // log sigmoid(k)
        float lc_v  = -softplus_f(kv);        // log(1 - sigmoid(k))
        float lv_v  = log_z + log_g_f(pv);
        size_t idx = (size_t)row * D_DIM + col;
        lc[idx] = lc_v;
        lv[idx] = lv_v;
    }
}

// =====================================================================
// Kernel 1: fused dual GEMM (k = x@Wz^T, pre = x@Wh^T) + gate math.
// Inputs pre-converted to f16. Block = 256 threads (8 waves),
// block tile 128(M) x 64(N); each wave owns a 32x32 tile of BOTH outputs:
// 8 f32 accumulators via v_wmma_f32_16x16x32_f16, A fragments shared.
// Inner loop: 12 b128-class loads + 8 WMMAs per K=32 step.
// =====================================================================
__global__ __launch_bounds__(256)
void mingru_gemm_gates(const _Float16* __restrict__ x,
                       const _Float16* __restrict__ Wh,
                       const _Float16* __restrict__ Wz,
                       float* __restrict__ lc,
                       float* __restrict__ lv)
{
    const int lane = threadIdx.x & 31;
    const int w    = threadIdx.x >> 5;
    const int half = lane >> 4;   // 0 or 1
    const int l16  = lane & 15;
    const int wm = w & 3, wn = w >> 2;

    const int mbase = blockIdx.y * 128 + wm * 32;  // 2 mtiles of 16
    const int nbase = blockIdx.x * 64  + wn * 32;  // 2 ntiles of 16

    v8f k00 = {}, k01 = {}, k10 = {}, k11 = {};
    v8f p00 = {}, p01 = {}, p10 = {}, p11 = {};

    const _Float16* xr0  = x  + (size_t)(mbase + l16)        * D_DIM + half * 16;
    const _Float16* xr1  = x  + (size_t)(mbase + 16 + l16)   * D_DIM + half * 16;
    const _Float16* wzr0 = Wz + (size_t)(nbase + l16)        * D_DIM + half * 16;
    const _Float16* wzr1 = Wz + (size_t)(nbase + 16 + l16)   * D_DIM + half * 16;
    const _Float16* whr0 = Wh + (size_t)(nbase + l16)        * D_DIM + half * 16;
    const _Float16* whr1 = Wh + (size_t)(nbase + 16 + l16)   * D_DIM + half * 16;

    for (int k0 = 0; k0 < D_DIM; k0 += 32) {
        v16h a0  = load_a16(xr0 + k0);
        v16h a1  = load_a16(xr1 + k0);
        v16h bz0 = load_b16(wzr0 + k0);
        v16h bz1 = load_b16(wzr1 + k0);
        v16h bh0 = load_b16(whr0 + k0);
        v16h bh1 = load_b16(whr1 + k0);

        k00 = __builtin_amdgcn_wmma_f32_16x16x32_f16(false, a0, false, bz0, (short)0, k00, false, false);
        k01 = __builtin_amdgcn_wmma_f32_16x16x32_f16(false, a0, false, bz1, (short)0, k01, false, false);
        k10 = __builtin_amdgcn_wmma_f32_16x16x32_f16(false, a1, false, bz0, (short)0, k10, false, false);
        k11 = __builtin_amdgcn_wmma_f32_16x16x32_f16(false, a1, false, bz1, (short)0, k11, false, false);
        p00 = __builtin_amdgcn_wmma_f32_16x16x32_f16(false, a0, false, bh0, (short)0, p00, false, false);
        p01 = __builtin_amdgcn_wmma_f32_16x16x32_f16(false, a0, false, bh1, (short)0, p01, false, false);
        p10 = __builtin_amdgcn_wmma_f32_16x16x32_f16(false, a1, false, bh0, (short)0, p10, false, false);
        p11 = __builtin_amdgcn_wmma_f32_16x16x32_f16(false, a1, false, bh1, (short)0, p11, false, false);
    }

    store_gate_tile(k00, p00, mbase +  0, nbase +  0, half, l16, lc, lv);
    store_gate_tile(k01, p01, mbase +  0, nbase + 16, half, l16, lc, lv);
    store_gate_tile(k10, p10, mbase + 16, nbase +  0, half, l16, lc, lv);
    store_gate_tile(k11, p11, mbase + 16, nbase + 16, half, l16, lc, lv);
}

// =====================================================================
// Kernel 2 (pass A): independent in-chunk scans, in place.
//   lv[idx] <- local log-scan state (seeded from -inf within chunk)
//   lc[idx] <- running cumsum of log_coeffs within chunk
// Coalesced: one block = 256 consecutive d for fixed (b, chunk).
// =====================================================================
__global__ __launch_bounds__(256)
void mingru_scan_chunks(float* __restrict__ lc, float* __restrict__ lv,
                        float* __restrict__ endLocal, float* __restrict__ endCum)
{
    const int d = blockIdx.x * 256 + threadIdx.x;  // grid.x = D/256
    const int c = blockIdx.y;                      // chunk
    const int b = blockIdx.z;                      // batch
    size_t base = ((size_t)b * T_SEQ + (size_t)c * CHUNK_LEN) * D_DIM + d;

    float state = 0.f;
    float cum   = 0.f;
    for (int i = 0; i < CHUNK_LEN; ++i) {
        size_t idx = base + (size_t)i * D_DIM;
        float a = lc[idx];
        float v = lv[idx];
        cum += a;
        state = (i == 0) ? v : logaddexp_f(state + a, v);
        lv[idx] = state;
        lc[idx] = cum;
    }
    int ch = b * D_DIM + d;
    endLocal[(size_t)ch * CHUNKS + c] = state;
    endCum  [(size_t)ch * CHUNKS + c] = cum;
}

// =====================================================================
// Kernel 3 (pass B): sequential scan over the 64 chunk carries per
// channel, seeded with log_g(h0[b,d]). One thread per (b,d) channel.
// =====================================================================
__global__ __launch_bounds__(256)
void mingru_scan_carries(const float* __restrict__ endLocal,
                         const float* __restrict__ endCum,
                         const float* __restrict__ h0,
                         float* __restrict__ carryIn)
{
    int ch = blockIdx.x * blockDim.x + threadIdx.x;  // B*D threads
    float carry = log_g_f(h0[ch]);                   // h0 is [B,1,D] flat = [B*D]
    size_t base = (size_t)ch * CHUNKS;
    for (int c = 0; c < CHUNKS; ++c) {
        carryIn[base + c] = carry;
        carry = logaddexp_f(endCum[base + c] + carry, endLocal[base + c]);
    }
}

// =====================================================================
// Kernel 4 (pass C): apply carries, exponentiate, write h.
//   h[b,t,d] = exp( logaddexp(local_t, cumA_t + carryIn[chunk(t)]) )
// =====================================================================
__global__ __launch_bounds__(256)
void mingru_apply(const float* __restrict__ lc, const float* __restrict__ lv,
                  const float* __restrict__ carryIn, float* __restrict__ out)
{
    size_t idx = (size_t)blockIdx.x * blockDim.x + threadIdx.x;
    int d = (int)(idx % D_DIM);
    size_t rt = idx / D_DIM;
    int t = (int)(rt % T_SEQ);
    int b = (int)(rt / T_SEQ);
    int c = t / CHUNK_LEN;
    int ch = b * D_DIM + d;
    float carry = carryIn[(size_t)ch * CHUNKS + c];
    out[idx] = __expf(logaddexp_f(lv[idx], lc[idx] + carry));
}

// =====================================================================
// Launch: inputs in setup_inputs() order: x, h0, W_h, W_z (all fp32).
// Workspace: lc(64MB) | lv(64MB) | endLocal|endCum|carryIn (1MB ea)
//            | xh(32MB f16) | Wh16(2MB) | Wz16(2MB)
// =====================================================================
extern "C" void kernel_launch(void* const* d_in, const int* in_sizes, int n_in,
                              void* d_out, int out_size, void* d_ws, size_t ws_size,
                              hipStream_t stream)
{
    const float* x  = (const float*)d_in[0];
    const float* h0 = (const float*)d_in[1];
    const float* Wh = (const float*)d_in[2];
    const float* Wz = (const float*)d_in[3];
    float* out = (float*)d_out;

    const size_t NELEM = (size_t)M_TOT * D_DIM;             // 16M
    const size_t WELEM = (size_t)D_DIM * D_DIM;             // 1M
    const size_t NCARR = (size_t)B_SZ * D_DIM * CHUNKS;

    float* lc       = (float*)d_ws;
    float* lv       = lc + NELEM;
    float* endLocal = lv + NELEM;
    float* endCum   = endLocal + NCARR;
    float* carryIn  = endCum + NCARR;
    _Float16* xh    = (_Float16*)(carryIn + NCARR);
    _Float16* Wh16  = xh + NELEM;
    _Float16* Wz16  = Wh16 + WELEM;

    // 0) one-shot fp32 -> f16 conversion (removes all in-loop cvt + halves GEMM bytes)
    cvt_f32_to_f16<<<(unsigned)(NELEM / 4 / 256), 256, 0, stream>>>(x,  xh);
    cvt_f32_to_f16<<<(unsigned)(WELEM / 4 / 256), 256, 0, stream>>>(Wh, Wh16);
    cvt_f32_to_f16<<<(unsigned)(WELEM / 4 / 256), 256, 0, stream>>>(Wz, Wz16);

    // 1) fused GEMMs + gate transforms
    dim3 g1(D_DIM / 64, M_TOT / 128);   // (16, 128)
    mingru_gemm_gates<<<g1, 256, 0, stream>>>(xh, Wh16, Wz16, lc, lv);

    // 2) parallel in-chunk scans
    dim3 g2(D_DIM / 256, CHUNKS, B_SZ); // (4, 64, 4)
    mingru_scan_chunks<<<g2, 256, 0, stream>>>(lc, lv, endLocal, endCum);

    // 3) carry scan per channel
    mingru_scan_carries<<<(B_SZ * D_DIM) / 256, 256, 0, stream>>>(endLocal, endCum, h0, carryIn);

    // 4) apply carries + exp -> output
    mingru_apply<<<(unsigned)(NELEM / 256), 256, 0, stream>>>(lc, lv, carryIn, out);
}
